// Attention_9457517985916
// MI455X (gfx1250) — matrix operand order repaired and checked by reading, not compiled, not run on hardware
//
#include <hip/hip_runtime.h>
#include <hip/hip_bf16.h>

// ---------------- problem constants ----------------
#define BB   128
#define NN   196
#define RNNC 2048
#define ATTC 1024

typedef __bf16 bf16_t;
typedef __attribute__((ext_vector_type(16))) __bf16 v16bf;
typedef __attribute__((ext_vector_type(8)))  float  v8f;

// ============================================================
// Weight transpose + f32 -> bf16 convert:  W[K][N] -> WT[N][K]
// ============================================================
__global__ __launch_bounds__(256) void transpose_cvt_kernel(
    const float* __restrict__ W, bf16_t* __restrict__ WT, int K, int N)
{
  __shared__ float t[32][33];
  const int n0 = blockIdx.x * 32, k0 = blockIdx.y * 32;
  const int tx = threadIdx.x, ty = threadIdx.y;
#pragma unroll
  for (int i = 0; i < 32; i += 8)
    t[ty + i][tx] = W[(size_t)(k0 + ty + i) * N + n0 + tx];
  __syncthreads();
#pragma unroll
  for (int i = 0; i < 32; i += 8)
    WT[(size_t)(n0 + ty + i) * K + k0 + tx] = (bf16_t)t[tx][ty + i];
}

// ============================================================
// Generic GEMM: C[M,N] = A[M,K](f32, cvt->bf16 on the fly) x
//               WT[N,K](bf16) + bias[N]
// Workgroup tile: 64 rows x 256 cols; 8 waves arranged 4(M) x 2(N);
// each wave: 16 rows x 128 cols = 8 WMMA subtiles (64 acc VGPRs).
// grid = (M/64, N/256), block = 256.  v_wmma_f32_16x16x32_bf16.
// sched_group_barrier forces [20 VMEM reads][8 cvt][8 WMMA] per
// k-step so B-fragment loads drain with partial waits.
// ============================================================
__global__ __launch_bounds__(256) void gemm_bf16_kernel(
    const float* __restrict__ A, const bf16_t* __restrict__ BT,
    const float* __restrict__ bias, float* __restrict__ C, int K, int N)
{
  const int tid  = threadIdx.x;
  const int wave = tid >> 5;
  const int mrow = wave & 3;           // 0..3  -> M sub-block
  const int ncol = wave >> 2;          // 0..1  -> N sub-block
  const int lane = tid & 31;
  const int half = lane >> 4;          // 0: lanes 0-15, 1: lanes 16-31
  const int r    = lane & 15;
  const int m0   = blockIdx.x * 64 + mrow * 16;
  const int nw   = blockIdx.y * 256 + ncol * 128;

  const float* __restrict__ arow = A + (size_t)(m0 + r) * K;

  // Hoisted B-fragment row bases (column n, K-offset half*16).
  const bf16_t* bptr[8];
#pragma unroll
  for (int t = 0; t < 8; ++t)
    bptr[t] = BT + (size_t)(nw + t * 16 + r) * K + half * 16;

  v8f acc[8];
#pragma unroll
  for (int t = 0; t < 8; ++t)
#pragma unroll
    for (int v = 0; v < 8; ++v) acc[t][v] = 0.0f;

  for (int k0 = 0; k0 < K; k0 += 32) {
    // Prefetch the A stream eight iterations ahead (speculative).
    __builtin_prefetch((const void*)(arow + k0 + 256), 0, 1);

    // A fragment (16x32 bf16), contiguous K layout (matches B):
    //  lane<16 : K = {k0..k0+15}
    //  lane>=16: K = {k0+16..k0+31}
    const float* p0 = arow + k0 + half * 16;
    const float* p1 = arow + k0 + half * 16 + 8;
    v16bf a;
#pragma unroll
    for (int i = 0; i < 8; ++i) {
      a[i]     = (bf16_t)p0[i];
      a[8 + i] = (bf16_t)p1[i];
    }

    // Batch-load all 8 B fragments (32B contiguous each).
    v16bf bfrag[8];
#pragma unroll
    for (int t = 0; t < 8; ++t)
      bfrag[t] = *(const v16bf*)(bptr[t] + k0);

#pragma unroll
    for (int t = 0; t < 8; ++t)
      acc[t] = __builtin_amdgcn_wmma_f32_16x16x32_bf16(
          false, a, false, bfrag[t], (short)0, acc[t], false, false);

    // Enforce: all VMEM reads first, then cvt VALU, then the WMMA burst.
    __builtin_amdgcn_sched_group_barrier(0x020, 20, 0);  // VMEM reads
    __builtin_amdgcn_sched_group_barrier(0x002, 8, 0);   // VALU (cvt_pk)
    __builtin_amdgcn_sched_group_barrier(0x008, 8, 0);   // MFMA/WMMA
  }

#pragma unroll
  for (int t = 0; t < 8; ++t) {
    const int n = nw + t * 16 + r;
    const float bv = bias ? bias[n] : 0.0f;
#pragma unroll
    for (int v = 0; v < 8; ++v)
      C[(size_t)(m0 + v + 8 * half) * N + n] = acc[t][v] + bv;
  }
}

// ============================================================
// score[m] = sum_a tanh(base[b,a] (+ base2[b,a]) + sgn*rows[m,a]) * wa[a] + wab
// b = m / rows_per_batch.  One 256-thread block per row m.
// ============================================================
__global__ __launch_bounds__(256) void score_kernel(
    const float* __restrict__ base, const float* __restrict__ base2,
    const float* __restrict__ rows, const float* __restrict__ wa,
    const float* __restrict__ wab, float sgn, float* __restrict__ score, int rpb)
{
  const int m = blockIdx.x;
  const int b = m / rpb;
  const float* br = base + (size_t)b * ATTC;
  const float* b2 = base2 ? base2 + (size_t)b * ATTC : nullptr;
  const float* rr = rows + (size_t)m * ATTC;
  float p = 0.0f;
  for (int a = threadIdx.x; a < ATTC; a += 256) {
    float x = br[a] + (b2 ? b2[a] : 0.0f) + sgn * rr[a];
    p += tanhf(x) * wa[a];
  }
  __shared__ float red[256];
  red[threadIdx.x] = p;
  __syncthreads();
  for (int s = 128; s > 0; s >>= 1) {
    if (threadIdx.x < s) red[threadIdx.x] += red[threadIdx.x + s];
    __syncthreads();
  }
  if (threadIdx.x == 0) score[m] = red[0] + wab[0];
}

// softmax over n rows per batch (block per b)
__global__ __launch_bounds__(256) void softmax_kernel(
    const float* __restrict__ score, float* __restrict__ w, int n)
{
  const int b = blockIdx.x, tid = threadIdx.x;
  const float* s = score + (size_t)b * n;
  __shared__ float red[256];
  float mx = -3.4e38f;
  for (int i = tid; i < n; i += 256) mx = fmaxf(mx, s[i]);
  red[tid] = mx; __syncthreads();
  for (int st = 128; st > 0; st >>= 1) {
    if (tid < st) red[tid] = fmaxf(red[tid], red[tid + st]);
    __syncthreads();
  }
  const float M = red[0];
  __syncthreads();
  float sum = 0.0f;
  for (int i = tid; i < n; i += 256) sum += expf(s[i] - M);
  red[tid] = sum; __syncthreads();
  for (int st = 128; st > 0; st >>= 1) {
    if (tid < st) red[tid] += red[tid + st];
    __syncthreads();
  }
  const float inv = 1.0f / red[0];
  float* o = w + (size_t)b * n;
  for (int i = tid; i < n; i += 256) o[i] = expf(s[i] - M) * inv;
}

// K=2 softmax with optional mask2=[1,cm] fold-in (mask*renorm == pre-scale e1)
__global__ void att2_kernel(const float* __restrict__ score,
                            const int* __restrict__ mask, float* __restrict__ w)
{
  const int b = threadIdx.x;
  float s0 = score[2 * b], s1 = score[2 * b + 1];
  float m = fmaxf(s0, s1);
  float e0 = expf(s0 - m), e1 = expf(s1 - m);
  if (mask) e1 *= (mask[b] > 0) ? 1.0f : 0.0f;
  float inv = 1.0f / (e0 + e1);
  w[2 * b] = e0 * inv;
  w[2 * b + 1] = e1 * inv;
}

// out[b,d] = (res? res[b,d] : 0) + sgn * sum_n w[b,n]*feats[b,n,d]
__global__ __launch_bounds__(256) void wsum_kernel(
    const float* __restrict__ w, const float* __restrict__ feats,
    const float* __restrict__ res, float sgn, float* __restrict__ out)
{
  const int b = blockIdx.x;
  const int d = blockIdx.y * 256 + threadIdx.x;
  const float* wb = w + (size_t)b * NN;
  const float* fb = feats + (size_t)b * NN * RNNC + d;
  float acc = 0.0f;
  for (int n = 0; n < NN; ++n) acc += wb[n] * fb[(size_t)n * RNNC];
  float o = sgn * acc + (res ? res[(size_t)b * RNNC + d] : 0.0f);
  out[(size_t)b * RNNC + d] = o;
}

// dst[2b,:]=s0[b,:]; dst[2b+1,:]=s1[b,:]
__global__ void pack2_kernel(float* __restrict__ dst, const float* __restrict__ s0,
                             const float* __restrict__ s1)
{
  size_t i = (size_t)blockIdx.x * 256 + threadIdx.x;  // over B*RNN
  size_t b = i / RNNC, d = i % RNNC;
  dst[(2 * b) * RNNC + d]     = s0[i];
  dst[(2 * b + 1) * RNNC + d] = s1[i];
}

// out[b,d] = w[2b]*x[b,d] + w[2b+1]*y[b,d]
__global__ void mix2_kernel(float* __restrict__ out, const float* __restrict__ w2,
                            const float* __restrict__ x, const float* __restrict__ y)
{
  size_t i = (size_t)blockIdx.x * 256 + threadIdx.x;
  size_t b = i / RNNC;
  out[i] = w2[2 * b] * x[i] + w2[2 * b + 1] * y[i];
}

__global__ void add_kernel(float* __restrict__ o, const float* __restrict__ a,
                           const float* __restrict__ b)
{
  size_t i = (size_t)blockIdx.x * 256 + threadIdx.x;
  o[i] = a[i] + b[i];
}

// glu_in[b,0:RNN]=outp[b,:]; glu_in[b,RNN:2RNN]=h[b,:]
__global__ void packglu_kernel(float* __restrict__ dst, const float* __restrict__ outp,
                               const float* __restrict__ h)
{
  size_t i = (size_t)blockIdx.x * 256 + threadIdx.x;  // over B*RNN
  size_t b = i / RNNC, d = i % RNNC;
  dst[b * (2 * RNNC) + d]        = outp[i];
  dst[b * (2 * RNNC) + RNNC + d] = h[i];
}

// x[b,d] = z[b,d] * sigmoid(z[b,RNN+d])
__global__ void glu_act_kernel(const float* __restrict__ z, float* __restrict__ x)
{
  size_t i = (size_t)blockIdx.x * 256 + threadIdx.x;  // over B*RNN
  size_t b = i / RNNC, d = i % RNNC;
  float aa = z[b * (2 * RNNC) + d];
  float bb = z[b * (2 * RNNC) + RNNC + d];
  x[i] = aa * (1.0f / (1.0f + expf(-bb)));
}

// ============================================================
extern "C" void kernel_launch(void* const* d_in, const int* in_sizes, int n_in,
                              void* d_out, int out_size, void* d_ws, size_t ws_size,
                              hipStream_t stream)
{
  (void)in_sizes; (void)n_in; (void)out_size; (void)ws_size;
  const float* h       = (const float*)d_in[0];
  const float* roi     = (const float*)d_in[1];   // [B,N,RNN]
  const float* proi    = (const float*)d_in[2];   // [B,N,ATT]
  const int*   mask    = (const int*)  d_in[3];
  const float* context = (const float*)d_in[4];
  const float* whg_w = (const float*)d_in[5],  *whg_b = (const float*)d_in[6];
  const float* wag_w = (const float*)d_in[7],  *wag_b = (const float*)d_in[8];
  const float* whc_w = (const float*)d_in[9],  *whc_b = (const float*)d_in[10];
  const float* wvc_w = (const float*)d_in[11], *wvc_b = (const float*)d_in[12];
  const float* wac_w = (const float*)d_in[13], *wac_b = (const float*)d_in[14];
  const float* whp_w = (const float*)d_in[15], *whp_b = (const float*)d_in[16];
  const float* wvp_w = (const float*)d_in[17], *wvp_b = (const float*)d_in[18];
  const float* wap_w = (const float*)d_in[19], *wap_b = (const float*)d_in[20];
  const float* wvo_w = (const float*)d_in[21], *wvo_b = (const float*)d_in[22];
  const float* who_w = (const float*)d_in[23], *who_b = (const float*)d_in[24];
  const float* wao_w = (const float*)d_in[25], *wao_b = (const float*)d_in[26];
  const float* glu_w = (const float*)d_in[27], *glu_b = (const float*)d_in[28];

  // ---- output slots (x, output, gw, cw, pw, ow) ----
  float* out_x      = (float*)d_out;
  float* out_output = out_x + (size_t)BB * RNNC;              // 262144
  float* gw_out     = out_output + (size_t)BB * RNNC;         // 524288
  float* cw_out     = gw_out + (size_t)BB * NN;               // 549376
  float* pw_out     = cw_out + (size_t)BB * 2;                // 549632
  float* ow_out     = pw_out + (size_t)BB * NN;               // 574720

  // ---- workspace carve-up ----
  char* ws = (char*)d_ws;
  size_t off = 0;
  auto carve = [&](size_t bytes) { void* p = ws + off; off += (bytes + 255) & ~(size_t)255; return p; };
  const size_t WT_BYTES = (size_t)ATTC * RNNC * sizeof(bf16_t);   // 4 MB each
  bf16_t* whgT = (bf16_t*)carve(WT_BYTES);
  bf16_t* whcT = (bf16_t*)carve(WT_BYTES);
  bf16_t* whpT = (bf16_t*)carve(WT_BYTES);
  bf16_t* whoT = (bf16_t*)carve(WT_BYTES);
  bf16_t* wvcT = (bf16_t*)carve(WT_BYTES);
  bf16_t* wvpT = (bf16_t*)carve(WT_BYTES);
  bf16_t* wvoT = (bf16_t*)carve(WT_BYTES);
  bf16_t* gluT = (bf16_t*)carve((size_t)4096 * 4096 * sizeof(bf16_t));  // 32 MB
  float* rproj   = (float*)carve((size_t)BB * NN * ATTC * sizeof(float)); // 98 MB
  float* hprojG  = (float*)carve((size_t)BB * ATTC * sizeof(float));
  float* hprojC  = (float*)carve((size_t)BB * ATTC * sizeof(float));
  float* hprojP  = (float*)carve((size_t)BB * ATTC * sizeof(float));
  float* hprojO  = (float*)carve((size_t)BB * ATTC * sizeof(float));
  float* cproj   = (float*)carve((size_t)BB * ATTC * sizeof(float));
  float* qP      = (float*)carve((size_t)BB * ATTC * sizeof(float));
  float* score_g = (float*)carve((size_t)BB * NN * sizeof(float));
  float* score_p = (float*)carve((size_t)BB * NN * sizeof(float));
  float* score_c = (float*)carve((size_t)BB * 2 * sizeof(float));
  float* score_o = (float*)carve((size_t)BB * 2 * sizeof(float));
  float* global_out  = (float*)carve((size_t)BB * RNNC * sizeof(float));
  float* context_out = (float*)carve((size_t)BB * RNNC * sizeof(float));
  float* comp_out    = (float*)carve((size_t)BB * RNNC * sizeof(float));
  float* roi2     = (float*)carve((size_t)BB * 2 * RNNC * sizeof(float));
  float* off2     = (float*)carve((size_t)BB * 2 * RNNC * sizeof(float));
  float* pfeats_c = (float*)carve((size_t)BB * 2 * ATTC * sizeof(float));
  float* pfeats_o = (float*)carve((size_t)BB * 2 * ATTC * sizeof(float));
  float* glu_in   = (float*)carve((size_t)BB * 2 * RNNC * sizeof(float));
  float* zbuf     = (float*)carve((size_t)BB * 2 * RNNC * sizeof(float));

  const dim3 tb(32, 8);
  const dim3 tg(ATTC / 32, RNNC / 32);        // N/32 x K/32
  const int EW = (BB * RNNC) / 256;           // 1024 blocks for elementwise

  // ---- 0. weight transpose + bf16 convert ----
  transpose_cvt_kernel<<<tg, tb, 0, stream>>>(whg_w, whgT, RNNC, ATTC);
  transpose_cvt_kernel<<<tg, tb, 0, stream>>>(whc_w, whcT, RNNC, ATTC);
  transpose_cvt_kernel<<<tg, tb, 0, stream>>>(whp_w, whpT, RNNC, ATTC);
  transpose_cvt_kernel<<<tg, tb, 0, stream>>>(who_w, whoT, RNNC, ATTC);
  transpose_cvt_kernel<<<tg, tb, 0, stream>>>(wvc_w, wvcT, RNNC, ATTC);
  transpose_cvt_kernel<<<tg, tb, 0, stream>>>(wvp_w, wvpT, RNNC, ATTC);
  transpose_cvt_kernel<<<tg, tb, 0, stream>>>(wvo_w, wvoT, RNNC, ATTC);
  transpose_cvt_kernel<<<dim3(4096 / 32, 4096 / 32), tb, 0, stream>>>(glu_w, gluT, 4096, 4096);

  // ---- 1. projections (WMMA) ----  grid = (M/64, N/256)
  gemm_bf16_kernel<<<dim3(BB / 64, ATTC / 256), 256, 0, stream>>>(h, whgT, whg_b, hprojG, RNNC, ATTC);
  gemm_bf16_kernel<<<dim3(BB / 64, ATTC / 256), 256, 0, stream>>>(h, whcT, whc_b, hprojC, RNNC, ATTC);
  gemm_bf16_kernel<<<dim3(BB / 64, ATTC / 256), 256, 0, stream>>>(h, whpT, whp_b, hprojP, RNNC, ATTC);
  gemm_bf16_kernel<<<dim3(BB / 64, ATTC / 256), 256, 0, stream>>>(h, whoT, who_b, hprojO, RNNC, ATTC);
  // big GEMM (105 GFLOP): rproj = roi_feats @ wvp (no bias; folded into cproj)
  gemm_bf16_kernel<<<dim3((BB * NN) / 64, ATTC / 256), 256, 0, stream>>>(roi, wvpT, (const float*)nullptr, rproj, RNNC, ATTC);

  // ---- 2. GlobalAttention ----
  score_kernel<<<BB * NN, 256, 0, stream>>>(hprojG, (const float*)nullptr, proi, wag_w, wag_b, 1.0f, score_g, NN);
  softmax_kernel<<<BB, 256, 0, stream>>>(score_g, gw_out, NN);
  wsum_kernel<<<dim3(BB, RNNC / 256), 256, 0, stream>>>(gw_out, roi, (const float*)nullptr, 1.0f, global_out);

  // ---- 3. ContextAttention (K=2, masked) ----
  pack2_kernel<<<EW, 256, 0, stream>>>(roi2, global_out, context);
  gemm_bf16_kernel<<<dim3((2 * BB) / 64, ATTC / 256), 256, 0, stream>>>(roi2, wvcT, wvc_b, pfeats_c, RNNC, ATTC);
  score_kernel<<<2 * BB, 256, 0, stream>>>(hprojC, (const float*)nullptr, pfeats_c, wac_w, wac_b, 1.0f, score_c, 2);
  att2_kernel<<<1, BB, 0, stream>>>(score_c, mask, cw_out);
  mix2_kernel<<<EW, 256, 0, stream>>>(context_out, cw_out, global_out, context);

  // ---- 4. CompositionAttention via rproj refactor ----
  gemm_bf16_kernel<<<dim3(BB / 64, ATTC / 256), 256, 0, stream>>>(context_out, wvpT, wvp_b, cproj, RNNC, ATTC);
  add_kernel<<<(BB * ATTC) / 256, 256, 0, stream>>>(qP, hprojP, cproj);
  // score_p[b,n] = sum_a tanh(qP[b,a] - rproj[b,n,a]) * wap + wap_b
  score_kernel<<<BB * NN, 256, 0, stream>>>(qP, (const float*)nullptr, rproj, wap_w, wap_b, -1.0f, score_p, NN);
  softmax_kernel<<<BB, 256, 0, stream>>>(score_p, pw_out, NN);
  // comp_out = context_out - sum_n pw*roi   (sum pw == 1)
  wsum_kernel<<<dim3(BB, RNNC / 256), 256, 0, stream>>>(pw_out, roi, context_out, -1.0f, comp_out);

  // ---- 5. OutputAttention (K=2) ----
  pack2_kernel<<<EW, 256, 0, stream>>>(off2, global_out, comp_out);
  gemm_bf16_kernel<<<dim3((2 * BB) / 64, ATTC / 256), 256, 0, stream>>>(off2, wvoT, wvo_b, pfeats_o, RNNC, ATTC);
  score_kernel<<<2 * BB, 256, 0, stream>>>(hprojO, (const float*)nullptr, pfeats_o, wao_w, wao_b, 1.0f, score_o, 2);
  att2_kernel<<<1, BB, 0, stream>>>(score_o, (const int*)nullptr, ow_out);
  mix2_kernel<<<EW, 256, 0, stream>>>(out_output, ow_out, global_out, comp_out);

  // ---- 6. GLU ----
  packglu_kernel<<<EW, 256, 0, stream>>>(glu_in, out_output, h);
  gemm_bf16_kernel<<<dim3(BB / 64, 4096 / 256), 256, 0, stream>>>(glu_in, gluT, glu_b, zbuf, 4096, 4096);
  glu_act_kernel<<<EW, 256, 0, stream>>>(zbuf, out_x);
}
